// SwitchFeedForward_5712306503876
// MI455X (gfx1250) — matrix operand order repaired and checked by reading, not compiled, not run on hardware
//
#include <hip/hip_runtime.h>
#include <math.h>

typedef __bf16 bf16;
typedef __attribute__((ext_vector_type(16))) __bf16 v16bf;
typedef __attribute__((ext_vector_type(8)))  float  v8f;

#define WMMA_BF16(A, B, C) \
  __builtin_amdgcn_wmma_f32_16x16x32_bf16(false, (A), false, (B), (short)0, (C), false, false)

static constexpr int Bn = 32, Sn = 1024, Dn = 768, Fn = 3072, En = 8;
static constexpr int TM = 128, TN = 128, TK = 32;
static constexpr int LDK = 40;  // padded LDS row stride (80 B = 16B-aligned, conflict-free half-wave)

// ---- fragment loaders from LDS (row-contiguous K, 32 bf16 per row) ----
// A 16x32 bf16 (M x K): lanes 0-15 -> kb=0, lanes 16-31 -> kb=8;
//   VGPR0..3 = K kb..kb+7, VGPR4..7 = K kb+16..kb+23
__device__ __forceinline__ v16bf fragA(const bf16* rowp, int lane) {
  const int kb = (lane & 16) ? 8 : 0;
  union { v16bf v; unsigned u[8]; } f;
#pragma unroll
  for (int i = 0; i < 4; ++i) {
    f.u[i]     = *(const unsigned*)(rowp + kb + 2 * i);
    f.u[i + 4] = *(const unsigned*)(rowp + kb + 16 + 2 * i);
  }
  return f.v;
}
// B 32x16 bf16 (K x N): lanes 0-15 -> kb=0, lanes 16-31 -> kb=8;
//   VGPR0..3 = K kb..kb+7, VGPR4..7 = K kb+16..kb+23
__device__ __forceinline__ v16bf fragB(const bf16* rowp, int lane) {
  const int kb = (lane & 16) ? 8 : 0;
  union { v16bf v; unsigned u[8]; } f;
#pragma unroll
  for (int i = 0; i < 4; ++i) {
    f.u[i]     = *(const unsigned*)(rowp + kb + 2 * i);
    f.u[i + 4] = *(const unsigned*)(rowp + kb + 16 + 2 * i);
  }
  return f.v;
}

// Async global->LDS 16-byte copy (CDNA5 ASYNCcnt path); lptr must be a shared-memory pointer.
__device__ __forceinline__ void async_copy_b128(const void* gptr, void* lptr) {
  unsigned lds_off = (unsigned)(size_t)lptr;                 // flat shared addr[31:0] == LDS offset
  unsigned long long ga = (unsigned long long)(size_t)gptr;
  asm volatile("global_load_async_to_lds_b128 %0, %1, off"
               :: "v"(lds_off), "v"(ga) : "memory");
}
__device__ __forceinline__ void async_wait0() {
  asm volatile("s_wait_asynccnt 0x0" ::: "memory");
}

// ---------------- routing (tiny) ----------------
__global__ __launch_bounds__(32) void route_kernel(
    const float* __restrict__ rt, const float* __restrict__ sw,
    const float* __restrict__ sb, float* __restrict__ out_tail,
    int* __restrict__ routes, float* __restrict__ scales) {
  __shared__ float probs_s[Bn][En];
  __shared__ int   route_s[Bn];
  __shared__ float rpm_s[Bn];
  const int b = threadIdx.x;
  if (b < Bn) {
    float logit[En];
#pragma unroll
    for (int e = 0; e < En; ++e) logit[e] = sb[e];
    const float* r = rt + (size_t)b * 2 * Dn;  // routing_tokens[:,0]
    for (int d = 0; d < Dn; ++d) {
      float rv = r[d];
#pragma unroll
      for (int e = 0; e < En; ++e) logit[e] += rv * sw[d * En + e];
    }
    float mx = logit[0];
#pragma unroll
    for (int e = 1; e < En; ++e) mx = fmaxf(mx, logit[e]);
    float p[En], sum = 0.f;
#pragma unroll
    for (int e = 0; e < En; ++e) { p[e] = expf(logit[e] - mx); sum += p[e]; }
    const float inv = 1.f / sum;
    float best = -1.f; int bi = 0;
#pragma unroll
    for (int e = 0; e < En; ++e) {
      p[e] *= inv;
      probs_s[b][e] = p[e];
      if (p[e] > best) { best = p[e]; bi = e; }  // first max wins (strict >)
    }
    route_s[b] = bi; rpm_s[b] = best;
    routes[b] = bi;  scales[b] = best;
  }
  __syncthreads();
  if (b == 0) {
    float counts[En], psum[En];
#pragma unroll
    for (int e = 0; e < En; ++e) { counts[e] = 0.f; psum[e] = 0.f; }
    for (int i = 0; i < Bn; ++i) {
      counts[route_s[i]] += 1.f;
#pragma unroll
      for (int e = 0; e < En; ++e) psum[e] += probs_s[i][e];
    }
#pragma unroll
    for (int e = 0; e < En; ++e) { out_tail[e] = counts[e]; out_tail[En + e] = psum[e]; }
    out_tail[2 * En] = 0.f;  // n_dropped
    for (int i = 0; i < Bn; ++i) out_tail[2 * En + 1 + i] = rpm_s[i];
  }
}

// ---------------- GEMM1: h = gelu(x @ w1[e] + b1[e]), split-store bf16 hi/lo ----------------
__global__ __launch_bounds__(256) void ffn_gemm1(
    const float* __restrict__ X, const float* __restrict__ W1,
    const float* __restrict__ B1, const int* __restrict__ routes,
    bf16* __restrict__ Hhi, bf16* __restrict__ Hlo) {
  __shared__ __attribute__((aligned(16))) bf16 sAh[TM * LDK];
  __shared__ __attribute__((aligned(16))) bf16 sAl[TM * LDK];
  __shared__ __attribute__((aligned(16))) bf16 sBh[TN * LDK];
  __shared__ __attribute__((aligned(16))) bf16 sBl[TN * LDK];

  const int b = blockIdx.z;
  const int e = routes[b];
  const float* x  = X  + (size_t)b * Sn * Dn;
  const float* w  = W1 + (size_t)e * Dn * Fn;
  const float* bs = B1 + (size_t)e * Fn;

  const int row0 = blockIdx.x * TM;
  const int col0 = blockIdx.y * TN;
  const int tid  = threadIdx.x;
  const int lane = tid & 31;
  const int wv   = tid >> 5;   // 8 waves
  const int wm   = wv >> 1;    // 0..3 -> 32-row strip
  const int wn   = wv & 1;     // 0..1 -> 64-col strip

  v8f acc[2][4];
#pragma unroll
  for (int m = 0; m < 2; ++m)
#pragma unroll
    for (int n = 0; n < 4; ++n)
      acc[m][n] = (v8f){0.f, 0.f, 0.f, 0.f, 0.f, 0.f, 0.f, 0.f};

  for (int k0 = 0; k0 < Dn; k0 += TK) {
#pragma unroll
    for (int i = 0; i < 16; ++i) {           // A tile 128x32, fp32 -> bf16 hi/lo
      int idx = tid + i * 256;
      int r = idx >> 5, k = idx & 31;
      float v = x[(size_t)(row0 + r) * Dn + (k0 + k)];
      bf16 hi = (bf16)v;
      sAh[r * LDK + k] = hi;
      sAl[r * LDK + k] = (bf16)(v - (float)hi);
    }
#pragma unroll
    for (int i = 0; i < 16; ++i) {           // B tile 32x128, stored [n][k]
      int idx = tid + i * 256;
      int k = idx >> 7, n = idx & 127;
      float v = w[(size_t)(k0 + k) * Fn + (col0 + n)];
      bf16 hi = (bf16)v;
      sBh[n * LDK + k] = hi;
      sBl[n * LDK + k] = (bf16)(v - (float)hi);
    }
    __syncthreads();

    v16bf ah[2], al[2], bh[4], bl[4];
#pragma unroll
    for (int m = 0; m < 2; ++m) {
      int row = wm * 32 + m * 16 + (lane & 15);
      ah[m] = fragA(sAh + row * LDK, lane);
      al[m] = fragA(sAl + row * LDK, lane);
    }
#pragma unroll
    for (int n = 0; n < 4; ++n) {
      int cn = wn * 64 + n * 16 + (lane & 15);
      bh[n] = fragB(sBh + cn * LDK, lane);
      bl[n] = fragB(sBl + cn * LDK, lane);
    }
#pragma unroll
    for (int m = 0; m < 2; ++m)
#pragma unroll
      for (int n = 0; n < 4; ++n) {
        acc[m][n] = WMMA_BF16(ah[m], bh[n], acc[m][n]);
        acc[m][n] = WMMA_BF16(ah[m], bl[n], acc[m][n]);
        acc[m][n] = WMMA_BF16(al[m], bh[n], acc[m][n]);
      }
    __syncthreads();
  }

  // epilogue: bias + exact gelu, split into bf16 hi/lo planes
#pragma unroll
  for (int m = 0; m < 2; ++m) {
    int rbase = row0 + wm * 32 + m * 16 + ((lane >> 4) << 3);
#pragma unroll
    for (int n = 0; n < 4; ++n) {
      int col = col0 + wn * 64 + n * 16 + (lane & 15);
      float bv = bs[col];
#pragma unroll
      for (int r = 0; r < 8; ++r) {
        float v = acc[m][n][r] + bv;
        v = 0.5f * v * (1.0f + erff(v * 0.70710678118654752f));
        bf16 hi = (bf16)v;
        size_t off = (size_t)b * Sn * Fn + (size_t)(rbase + r) * Fn + col;
        Hhi[off] = hi;
        Hlo[off] = (bf16)(v - (float)hi);
      }
    }
  }
}

// ---------------- GEMM2: out = (h @ w2[e] + b2[e]) * route_prob_max ----------------
// A tiles (already bf16) staged with CDNA5 async global->LDS copies (ASYNCcnt);
// B tiles converted fp32->bf16 hi/lo in VALU concurrently.
__global__ __launch_bounds__(256) void ffn_gemm2(
    const bf16* __restrict__ Hhi, const bf16* __restrict__ Hlo,
    const float* __restrict__ W2, const float* __restrict__ B2,
    const int* __restrict__ routes, const float* __restrict__ scales,
    float* __restrict__ Out) {
  __shared__ __attribute__((aligned(16))) bf16 sAh[TM * LDK];
  __shared__ __attribute__((aligned(16))) bf16 sAl[TM * LDK];
  __shared__ __attribute__((aligned(16))) bf16 sBh[TN * LDK];
  __shared__ __attribute__((aligned(16))) bf16 sBl[TN * LDK];

  const int b = blockIdx.z;
  const int e = routes[b];
  const float scale = scales[b];
  const float* w  = W2 + (size_t)e * Fn * Dn;
  const float* bs = B2 + (size_t)e * Dn;

  const int row0 = blockIdx.x * TM;
  const int col0 = blockIdx.y * TN;
  const int tid  = threadIdx.x;
  const int lane = tid & 31;
  const int wv   = tid >> 5;
  const int wm   = wv >> 1;
  const int wn   = wv & 1;

  v8f acc[2][4];
#pragma unroll
  for (int m = 0; m < 2; ++m)
#pragma unroll
    for (int n = 0; n < 4; ++n)
      acc[m][n] = (v8f){0.f, 0.f, 0.f, 0.f, 0.f, 0.f, 0.f, 0.f};

  for (int k0 = 0; k0 < Fn; k0 += TK) {
    // A tile (128x32 bf16 per plane): pure byte copy -> async global->LDS.
    // 512 16-byte chunks per plane; each thread issues 2 per plane.
#pragma unroll
    for (int i = 0; i < 2; ++i) {
      int c = tid + i * 256;            // chunk id 0..511
      int r = c >> 2, kc = c & 3;       // row 0..127, 16B (8 bf16) chunk in row
      size_t goff = (size_t)b * Sn * Fn + (size_t)(row0 + r) * Fn + (size_t)(k0 + kc * 8);
      async_copy_b128(Hhi + goff, &sAh[r * LDK + kc * 8]);
      async_copy_b128(Hlo + goff, &sAl[r * LDK + kc * 8]);
    }
    // B tile 32x128 from w2 (F x D), fp32 -> bf16 hi/lo, stored [n][k]
#pragma unroll
    for (int i = 0; i < 16; ++i) {
      int idx = tid + i * 256;
      int k = idx >> 7, n = idx & 127;
      float v = w[(size_t)(k0 + k) * Dn + (col0 + n)];
      bf16 hi = (bf16)v;
      sBh[n * LDK + k] = hi;
      sBl[n * LDK + k] = (bf16)(v - (float)hi);
    }
    async_wait0();       // this wave's async LDS writes complete
    __syncthreads();     // all waves' writes visible

    v16bf ah[2], al[2], bh[4], bl[4];
#pragma unroll
    for (int m = 0; m < 2; ++m) {
      int row = wm * 32 + m * 16 + (lane & 15);
      ah[m] = fragA(sAh + row * LDK, lane);
      al[m] = fragA(sAl + row * LDK, lane);
    }
#pragma unroll
    for (int n = 0; n < 4; ++n) {
      int cn = wn * 64 + n * 16 + (lane & 15);
      bh[n] = fragB(sBh + cn * LDK, lane);
      bl[n] = fragB(sBl + cn * LDK, lane);
    }
#pragma unroll
    for (int m = 0; m < 2; ++m)
#pragma unroll
      for (int n = 0; n < 4; ++n) {
        acc[m][n] = WMMA_BF16(ah[m], bh[n], acc[m][n]);
        acc[m][n] = WMMA_BF16(ah[m], bl[n], acc[m][n]);
        acc[m][n] = WMMA_BF16(al[m], bh[n], acc[m][n]);
      }
    __syncthreads();
  }

#pragma unroll
  for (int m = 0; m < 2; ++m) {
    int rbase = row0 + wm * 32 + m * 16 + ((lane >> 4) << 3);
#pragma unroll
    for (int n = 0; n < 4; ++n) {
      int col = col0 + wn * 64 + n * 16 + (lane & 15);
      float bv = bs[col];
#pragma unroll
      for (int r = 0; r < 8; ++r) {
        float v = (acc[m][n][r] + bv) * scale;
        Out[(size_t)b * Sn * Dn + (size_t)(rbase + r) * Dn + col] = v;
      }
    }
  }
}

extern "C" void kernel_launch(void* const* d_in, const int* in_sizes, int n_in,
                              void* d_out, int out_size, void* d_ws, size_t ws_size,
                              hipStream_t stream) {
  (void)in_sizes; (void)n_in; (void)out_size; (void)ws_size;
  const float* x  = (const float*)d_in[0];
  const float* rt = (const float*)d_in[1];
  const float* sw = (const float*)d_in[2];
  const float* sb = (const float*)d_in[3];
  const float* w1 = (const float*)d_in[4];
  const float* b1 = (const float*)d_in[5];
  const float* w2 = (const float*)d_in[6];
  const float* b2 = (const float*)d_in[7];
  float* out = (float*)d_out;

  // workspace: h hi plane | h lo plane | routes | scales
  const size_t Hn = (size_t)Bn * Sn * Fn;
  bf16*  Hhi    = (bf16*)d_ws;
  bf16*  Hlo    = Hhi + Hn;
  int*   routes = (int*)((char*)d_ws + 4 * Hn);
  float* scales = (float*)((char*)d_ws + 4 * Hn + 128);
  float* out_tail = out + (size_t)Bn * Sn * Dn;

  route_kernel<<<1, 32, 0, stream>>>(rt, sw, sb, out_tail, routes, scales);
  ffn_gemm1<<<dim3(Sn / TM, Fn / TN, Bn), 256, 0, stream>>>(x, w1, b1, routes, Hhi, Hlo);
  ffn_gemm2<<<dim3(Sn / TM, Dn / TN, Bn), 256, 0, stream>>>(Hhi, Hlo, w2, b2, routes, scales, out);
}
